// RNN_LSTM_45251775430853
// MI455X (gfx1250) — compile-verified, matrix operand order checked by reading, not run on hardware
//
#include <hip/hip_runtime.h>

// ---------------------------------------------------------------------------
// LSTM final-hidden-state for MI455X (gfx1250, wave32, WMMA).
// B=128, T=1024, I=256, H=512.  8 independent workgroups (16 batch rows each),
// 1024 threads = 32 waves per WGP.  Per step:
//   gates[16,2048] = [x_t | h](16x768) @ W_cat^T   via v_wmma_f32_16x16x32_bf16
// c stays in VGPRs for all 1024 steps; h round-trips through LDS as bf16.
// ---------------------------------------------------------------------------

typedef __attribute__((ext_vector_type(16))) __bf16 v16bf;
typedef __attribute__((ext_vector_type(8)))  __bf16 v8bf;
typedef __attribute__((ext_vector_type(8)))  float  v8f;

#define B_    128
#define T_    1024
#define I_    256
#define H_    512
#define G_    2048          // 4*H
#define K_    768           // I + H
#define LDSTR 784           // padded LDS row stride in bf16 (rows stay 32B aligned)

union BF16x16 { v16bf v; v8bf h[2]; };

// Load a 16-element bf16 fragment honoring the CDNA5 16-bit operand layout:
// lanes 0-15 (sub=0): K = {k0..k0+7} U {k0+16..k0+23}
// lanes 16-31 (sub=1): K = {k0+8..k0+15} U {k0+24..k0+31}
__device__ __forceinline__ v16bf frag_ld(const __bf16* p, int sub) {
    BF16x16 r;
    r.h[0] = *(const v8bf*)(p + 8 * sub);
    r.h[1] = *(const v8bf*)(p + 16 + 8 * sub);
    return r.v;
}

__device__ __forceinline__ float sigmoidf_(float x) {
    return 1.0f / (1.0f + __expf(-x));
}
__device__ __forceinline__ float tanhf_(float x) {
    return 1.0f - 2.0f / (__expf(2.0f * x) + 1.0f);
}

// ---------------------------------------------------------------------------
// Prep: W_cat[n][k] bf16 row-major (k<256 -> W_ih, else W_hh); bias = bih+bhh.
// ---------------------------------------------------------------------------
__global__ void lstm_prep(const float* __restrict__ Wih, const float* __restrict__ Whh,
                          const float* __restrict__ bih, const float* __restrict__ bhh,
                          __bf16* __restrict__ Wc, float* __restrict__ bias) {
    int idx = blockIdx.x * blockDim.x + threadIdx.x;
    if (idx < G_ * K_) {
        int n = idx / K_;
        int k = idx - n * K_;
        float v = (k < I_) ? Wih[n * I_ + k] : Whh[n * H_ + (k - I_)];
        Wc[(size_t)n * K_ + k] = (__bf16)v;
    }
    if (idx < G_) bias[idx] = bih[idx] + bhh[idx];
}

// ---------------------------------------------------------------------------
// Sequential LSTM.  One workgroup per 16-row batch tile.  No inter-WG sync.
// ---------------------------------------------------------------------------
__global__ __launch_bounds__(1024, 1)
void lstm_seq(const float* __restrict__ x, const __bf16* __restrict__ Wc,
              const float* __restrict__ bias, float* __restrict__ out) {
    __shared__ __bf16 xh[16][LDSTR];   // cols [0,256): x_t ; [256,768): h (bf16)

    const int tid    = threadIdx.x;
    const int wave   = tid >> 5;       // 0..31
    const int lane   = tid & 31;
    const int lane_n = lane & 15;      // N (and A-row M) index within tile
    const int sub    = lane >> 4;      // K sub-chunk selector
    const int b0     = blockIdx.x * 16;
    const int hs     = wave * 16;      // hidden-slice base column for this wave

    // Per-lane gate bias (accumulator init); lane's column is fixed per tile.
    const float bi = bias[0 * H_ + hs + lane_n];
    const float bf = bias[1 * H_ + hs + lane_n];
    const float bg = bias[2 * H_ + hs + lane_n];
    const float bo = bias[3 * H_ + hs + lane_n];

    // B-matrix rows (row-major W_cat -> contiguous K per output column).
    const __bf16* Wr_i = Wc + (size_t)(0 * H_ + hs + lane_n) * K_;
    const __bf16* Wr_f = Wc + (size_t)(1 * H_ + hs + lane_n) * K_;
    const __bf16* Wr_g = Wc + (size_t)(2 * H_ + hs + lane_n) * K_;
    const __bf16* Wr_o = Wc + (size_t)(3 * H_ + hs + lane_n) * K_;

    // h0 = 0
    for (int e = tid; e < 16 * H_; e += 1024)
        xh[e >> 9][I_ + (e & 511)] = (__bf16)0.0f;
    // stage x(t=0): 64 threads per row, 4 floats each
    {
        const int m = tid >> 6, i4 = (tid & 63) * 4;
        const float4 xv = *(const float4*)(x + ((size_t)(b0 + m) * T_ + 0) * I_ + i4);
        xh[m][i4 + 0] = (__bf16)xv.x; xh[m][i4 + 1] = (__bf16)xv.y;
        xh[m][i4 + 2] = (__bf16)xv.z; xh[m][i4 + 3] = (__bf16)xv.w;
    }
    __syncthreads();

    float c[8];
    #pragma unroll
    for (int v = 0; v < 8; ++v) c[v] = 0.0f;

    const __bf16* arow = &xh[lane_n][0];

    for (int t = 0; t < T_; ++t) {
        v8f ai, af, ag, ao;
        #pragma unroll
        for (int v = 0; v < 8; ++v) { ai[v] = bi; af[v] = bf; ag[v] = bg; ao[v] = bo; }

        #pragma unroll 4
        for (int k0 = 0; k0 < K_; k0 += 32) {
            v16bf a  = frag_ld(arow + k0, sub);      // shared A across 4 gate tiles
            v16bf wi = frag_ld(Wr_i + k0, sub);
            v16bf wf = frag_ld(Wr_f + k0, sub);
            v16bf wg = frag_ld(Wr_g + k0, sub);
            v16bf wo = frag_ld(Wr_o + k0, sub);
            ai = __builtin_amdgcn_wmma_f32_16x16x32_bf16(false, a, false, wi, (short)0, ai, false, false);
            af = __builtin_amdgcn_wmma_f32_16x16x32_bf16(false, a, false, wf, (short)0, af, false, false);
            ag = __builtin_amdgcn_wmma_f32_16x16x32_bf16(false, a, false, wg, (short)0, ag, false, false);
            ao = __builtin_amdgcn_wmma_f32_16x16x32_bf16(false, a, false, wo, (short)0, ao, false, false);
        }

        // Elementwise cell update.  D layout: VGPR v -> row M = v + 8*sub.
        float hv[8];
        #pragma unroll
        for (int v = 0; v < 8; ++v) {
            float ig = sigmoidf_(ai[v]);
            float fg = sigmoidf_(af[v]);
            float gg = tanhf_(ag[v]);
            float og = sigmoidf_(ao[v]);
            c[v]  = fg * c[v] + ig * gg;
            hv[v] = og * tanhf_(c[v]);
        }

        __syncthreads();   // all waves finished reading xh for this step

        // write new h slice into LDS (bf16)
        #pragma unroll
        for (int v = 0; v < 8; ++v)
            xh[v + 8 * sub][I_ + hs + lane_n] = (__bf16)hv[v];

        if (t + 1 < T_) {
            // stage x(t+1)
            const int m = tid >> 6, i4 = (tid & 63) * 4;
            const float4 xv = *(const float4*)(x + ((size_t)(b0 + m) * T_ + (t + 1)) * I_ + i4);
            xh[m][i4 + 0] = (__bf16)xv.x; xh[m][i4 + 1] = (__bf16)xv.y;
            xh[m][i4 + 2] = (__bf16)xv.z; xh[m][i4 + 3] = (__bf16)xv.w;
        } else {
            // final step: emit h_T in fp32
            #pragma unroll
            for (int v = 0; v < 8; ++v)
                out[(size_t)(b0 + v + 8 * sub) * H_ + hs + lane_n] = hv[v];
        }
        __syncthreads();
    }
}

// ---------------------------------------------------------------------------
extern "C" void kernel_launch(void* const* d_in, const int* in_sizes, int n_in,
                              void* d_out, int out_size, void* d_ws, size_t ws_size,
                              hipStream_t stream) {
    const float* x   = (const float*)d_in[0];   // [B, T, I]
    const float* Wih = (const float*)d_in[1];   // [4H, I]
    const float* Whh = (const float*)d_in[2];   // [4H, H]
    const float* bih = (const float*)d_in[3];   // [4H]
    const float* bhh = (const float*)d_in[4];   // [4H]

    __bf16* Wc  = (__bf16*)d_ws;                                  // 2048*768*2 = 3 MB
    float*  bias = (float*)((char*)d_ws + (size_t)G_ * K_ * 2);   // +8 KB
    float*  out  = (float*)d_out;                                 // [B, H]

    const int pt = 256;
    const int pb = (G_ * K_ + pt - 1) / pt;
    lstm_prep<<<pb, pt, 0, stream>>>(Wih, Whh, bih, bhh, Wc, bias);
    lstm_seq<<<B_ / 16, 1024, 0, stream>>>(x, Wc, bias, out);
}
